// LLM_39814346834310
// MI455X (gfx1250) — matrix operand order repaired and checked by reading, not compiled, not run on hardware
//
#include <hip/hip_runtime.h>
#include <hip/hip_bf16.h>
#include <math.h>

// ---- model dims ----
#define Vv   65
#define Dd   384
#define NLl  12
#define Tt   512
#define Bb   32
#define DIi  768
#define DSs  16
#define DCc  4
#define DTRr 24
#define DFFf 1536
#define XPJ  (DTRr + 2 * DSs)   // 56
#define ROWS (Bb * Tt)          // 16384 (multiple of 64 -- GEMM M is never ragged)
#define EPSf 1e-5f

typedef __bf16 bf16_t;
typedef __attribute__((ext_vector_type(16))) __bf16 v16bf;
typedef __attribute__((ext_vector_type(8)))  float  v8f;

// ---------------- embedding ----------------
__global__ __launch_bounds__(256) void embed_kernel(
    const int* __restrict__ idx, const float* __restrict__ tok_emb,
    const float* __restrict__ pos_emb, float* __restrict__ x)
{
    int e = blockIdx.x * 256 + threadIdx.x;
    if (e >= ROWS * Dd) return;
    int d  = e % Dd;
    int bt = e / Dd;
    int t  = bt % Tt;
    int tok = idx[bt];
    x[e] = tok_emb[tok * Dd + d] + pos_emb[t * Dd + d];
}

// ---------------- layernorm: one wave (32 lanes) per row of D=384 ----------------
__global__ __launch_bounds__(256) void layernorm_kernel(
    const float* __restrict__ X, const float* __restrict__ g,
    const float* __restrict__ b, float* __restrict__ Y)
{
    int wave = threadIdx.x >> 5;
    int lane = threadIdx.x & 31;
    int row  = blockIdx.x * 8 + wave;
    if (row >= ROWS) return;
    const float* xr = X + (size_t)row * Dd;
    float s = 0.f, sq = 0.f;
    #pragma unroll
    for (int i = 0; i < Dd / 32; ++i) {
        float v = xr[lane + i * 32];
        s += v; sq += v * v;
    }
    #pragma unroll
    for (int off = 16; off > 0; off >>= 1) {
        s  += __shfl_xor(s,  off, 32);
        sq += __shfl_xor(sq, off, 32);
    }
    float mean = s * (1.f / Dd);
    float var  = sq * (1.f / Dd) - mean * mean;
    float rstd = rsqrtf(var + EPSf);
    float* yr = Y + (size_t)row * Dd;
    #pragma unroll
    for (int i = 0; i < Dd / 32; ++i) {
        int c = lane + i * 32;
        yr[c] = (xr[c] - mean) * rstd * g[c] + b[c];
    }
}

// ---------------- generic WMMA GEMM: C = act(A @ W^T + bias) + beta*C ----------------
// A: [M,K] row stride lda (f32). W: [N,K] row-major (f32). C: [M,N] row stride ldc.
// Block tile 64(M) x 128(N) x 32(K); 8 waves; each wave = 1 A frag x 4 B frags = 4 WMMA.
// act: 0=none 1=relu 2=softplus.  Requires: M % 64 == 0, lda/K even.
// RAGGED=false additionally requires K % 32 == 0 and N % 128 == 0.
template<bool RAGGED>
__global__ __launch_bounds__(256) void gemm_bf16_wmma(
    const float* __restrict__ A, const float* __restrict__ W,
    const float* __restrict__ bias, float* __restrict__ C,
    int M, int N, int K, int lda, int ldc, int act, int beta)
{
    // bf16 tiles as packed pairs; row stride 17 u32 = 34 bf16 (bank decorrelation)
    __shared__ unsigned Asu[64][17];
    __shared__ unsigned Wsu[128][17];

    const int tid  = threadIdx.x;
    const int lane = tid & 31;
    const int wave = tid >> 5;     // 0..7
    const int wm   = wave & 3;     // M sub-tile (16 rows)
    const int wn   = wave >> 2;    // N group (64 cols)
    const int m0   = blockIdx.y * 64;
    const int n0   = blockIdx.x * 128;
    const int l15  = lane & 15;
    const int hi   = lane >> 4;

    v8f acc0 = {}, acc1 = {}, acc2 = {}, acc3 = {};
    float2 areg[4], wreg[8];

    const int kTiles = (K + 31) / 32;

    auto load_tiles = [&](int k0) {
        #pragma unroll
        for (int e = 0; e < 4; ++e) {           // A: 64 rows x 16 pairs
            int pi = e * 256 + tid;
            int r  = pi >> 4, cp = pi & 15;
            int gc = k0 + 2 * cp;
            int gr = m0 + r;                    // always < M
            if constexpr (RAGGED) {
                bool okK = gc < K;              // K even -> pair fully in/out
                int gcc = okK ? gc : 0;
                float2 va = *(const float2*)(A + gr * lda + gcc);
                if (!okK) { va.x = 0.f; va.y = 0.f; }
                areg[e] = va;
            } else {
                areg[e] = *(const float2*)(A + gr * lda + gc);
            }
        }
        #pragma unroll
        for (int e = 0; e < 8; ++e) {           // W: 128 rows x 16 pairs
            int pi = e * 256 + tid;
            int r  = pi >> 4, cp = pi & 15;
            int gc = k0 + 2 * cp;
            int gn = n0 + r;
            if constexpr (RAGGED) {
                bool okK = gc < K;
                bool okW = okK && (gn < N);
                int gcc = okK ? gc : 0;
                int gnc = (gn < N) ? gn : (N - 1);
                float2 vw = *(const float2*)(W + gnc * K + gcc);
                if (!okW) { vw.x = 0.f; vw.y = 0.f; }
                wreg[e] = vw;
            } else {
                wreg[e] = *(const float2*)(W + gn * K + gc);
            }
        }
    };

    auto store_tiles = [&]() {
        #pragma unroll
        for (int e = 0; e < 4; ++e) {
            int pi = e * 256 + tid;
            int r = pi >> 4, cp = pi & 15;
            union { bf16_t h[2]; unsigned u; } pa;
            pa.h[0] = (bf16_t)areg[e].x; pa.h[1] = (bf16_t)areg[e].y;
            Asu[r][cp] = pa.u;
        }
        #pragma unroll
        for (int e = 0; e < 8; ++e) {
            int pi = e * 256 + tid;
            int r = pi >> 4, cp = pi & 15;
            union { bf16_t h[2]; unsigned u; } pw;
            pw.h[0] = (bf16_t)wreg[e].x; pw.h[1] = (bf16_t)wreg[e].y;
            Wsu[r][cp] = pw.u;
        }
    };

    const bf16_t* Asb = (const bf16_t*)Asu;  // row stride 34
    const bf16_t* Wsb = (const bf16_t*)Wsu;

    load_tiles(0);
    for (int kt = 0; kt < kTiles; ++kt) {
        __syncthreads();            // previous tile's fragment reads done
        store_tiles();              // waits on staged global loads
        __syncthreads();
        if (kt + 1 < kTiles) load_tiles((kt + 1) * 32);  // prefetch next tile

        // A fragment (ISA 7.12.2, 16-bit A 16x32, wave32)
        v16bf a;
        {
            int ra = (wm * 16 + l15) * 34;
            #pragma unroll
            for (int v = 0; v < 8; ++v) {
                int ka = (hi ? 8 : 0) + ((v < 4) ? (2 * v) : (16 + 2 * (v - 4)));
                a[2 * v]     = Asb[ra + ka];
                a[2 * v + 1] = Asb[ra + ka + 1];
            }
        }
        // 4 B fragments (16-bit B 32x16), 4 WMMAs sharing A
        #pragma unroll
        for (int j = 0; j < 4; ++j) {
            v16bf bf;
            int rb = (wn * 64 + j * 16 + l15) * 34;
            #pragma unroll
            for (int v = 0; v < 8; ++v) {
                int kb = (hi ? 8 : 0) + ((v < 4) ? (2 * v) : (16 + 2 * (v - 4)));
                bf[2 * v]     = Wsb[rb + kb];
                bf[2 * v + 1] = Wsb[rb + kb + 1];
            }
            v8f* accp = (j == 0) ? &acc0 : (j == 1) ? &acc1 : (j == 2) ? &acc2 : &acc3;
            *accp = __builtin_amdgcn_wmma_f32_16x16x32_bf16(false, a, false, bf, (short)0, *accp, false, false);
        }
    }

    // epilogue: C/D layout -> vgpr r: M = r + 8*hi, N = lane&15
    #pragma unroll
    for (int j = 0; j < 4; ++j) {
        const v8f& cc = (j == 0) ? acc0 : (j == 1) ? acc1 : (j == 2) ? acc2 : acc3;
        int gc = n0 + wn * 64 + j * 16 + l15;
        if (RAGGED && gc >= N) continue;
        float bv = bias ? bias[gc] : 0.f;
        #pragma unroll
        for (int r = 0; r < 8; ++r) {
            int grow = m0 + wm * 16 + hi * 8 + r;   // always < M
            float v = cc[r] + bv;
            if (act == 1) v = fmaxf(v, 0.f);
            else if (act == 2) v = (v > 20.f) ? v : __logf(1.f + __expf(v));
            float* cp = C + grow * ldc + gc;
            if (beta) v += *cp;
            *cp = v;
        }
    }
}

// ---------------- causal depthwise conv1d (k=4) + SiLU ----------------
__global__ __launch_bounds__(256) void conv_silu_kernel(
    const float* __restrict__ xz, const float* __restrict__ cw,
    const float* __restrict__ cb, float* __restrict__ xc)
{
    int e = blockIdx.x * 256 + threadIdx.x;
    if (e >= ROWS * DIi) return;
    int d  = e % DIi;
    int bt = e / DIi;
    int t  = bt % Tt;
    int b  = bt / Tt;
    float acc = cb[d];
    #pragma unroll
    for (int k = 0; k < DCc; ++k) {
        int tt = t - (DCc - 1) + k;
        if (tt >= 0)
            acc += xz[(b * Tt + tt) * (2 * DIi) + d] * cw[d * DCc + k];
    }
    xc[e] = acc / (1.f + __expf(-acc)); // SiLU
}

// ---------------- selective scan (sequential over T), fused D-skip + SiLU(z) gate ----------------
// B_t/C_t rows are block-uniform (address depends only on blockIdx.y and t):
// loads scalarize to SMEM / broadcast from L0 -- no LDS, no barriers in the serial loop.
__global__ __launch_bounds__(256) void scan_kernel(
    const float* __restrict__ xz,    // [ROWS, 2*DI] (z in second half)
    const float* __restrict__ xc,    // [ROWS, DI]
    const float* __restrict__ xdbl,  // [ROWS, 56] (dt|B|C)
    const float* __restrict__ dtb,   // [ROWS, DI] softplus'd dt
    const float* __restrict__ A_log, // [DI, DS]
    const float* __restrict__ Dp,    // [DI]
    float* __restrict__ y)           // [ROWS, DI]
{
    int tid = threadIdx.x;
    int d   = blockIdx.x * 256 + tid;   // channel
    int b   = blockIdx.y;               // batch

    float Ad[DSs];
    #pragma unroll
    for (int s = 0; s < DSs; ++s) Ad[s] = -__expf(A_log[d * DSs + s]);
    float dskip = Dp[d];
    float h[DSs];
    #pragma unroll
    for (int s = 0; s < DSs; ++s) h[s] = 0.f;

    for (int t = 0; t < Tt; ++t) {
        int rbase = b * Tt + t;
        const float* bc = xdbl + rbase * XPJ + DTRr;  // block-uniform address
        float dt_t = dtb[rbase * DIi + d];
        float x_t  = xc [rbase * DIi + d];
        float z_t  = xz [rbase * (2 * DIi) + DIi + d];
        float dtx  = dt_t * x_t;
        float acc  = 0.f;
        #pragma unroll
        for (int s = 0; s < DSs; ++s) {
            float dA = __expf(dt_t * Ad[s]);
            h[s] = dA * h[s] + dtx * bc[s];
            acc += h[s] * bc[DSs + s];
        }
        acc = (acc + x_t * dskip) * (z_t / (1.f + __expf(-z_t)));
        y[rbase * DIi + d] = acc;
    }
}

// ---------------- per-row cross-entropy from logits (one wave per row) ----------------
__global__ __launch_bounds__(256) void row_loss_kernel(
    const float* __restrict__ logits, const int* __restrict__ targets,
    float* __restrict__ rowloss)
{
    int wave = threadIdx.x >> 5;
    int lane = threadIdx.x & 31;
    int row  = blockIdx.x * 8 + wave;
    if (row >= ROWS) return;
    const float* lr = logits + (size_t)row * Vv;
    float m = -3.4e38f;
    for (int c = lane; c < Vv; c += 32) m = fmaxf(m, lr[c]);
    #pragma unroll
    for (int off = 16; off > 0; off >>= 1) m = fmaxf(m, __shfl_xor(m, off, 32));
    float s = 0.f;
    for (int c = lane; c < Vv; c += 32) s += __expf(lr[c] - m);
    #pragma unroll
    for (int off = 16; off > 0; off >>= 1) s += __shfl_xor(s, off, 32);
    if (lane == 0) {
        int tgt = targets[row];
        rowloss[row] = -(lr[tgt] - m - __logf(s));
    }
}

__global__ __launch_bounds__(256) void reduce_loss_kernel(
    const float* __restrict__ rowloss, float* __restrict__ out)
{
    __shared__ float sm[256];
    int tid = threadIdx.x;
    float s = 0.f;
    for (int i = tid; i < ROWS; i += 256) s += rowloss[i];
    sm[tid] = s;
    __syncthreads();
    #pragma unroll
    for (int off = 128; off > 0; off >>= 1) {
        if (tid < off) sm[tid] += sm[tid + off];
        __syncthreads();
    }
    if (tid == 0) out[0] = sm[0] * (1.f / ROWS);
}

// ---------------- launcher ----------------
static inline void run_gemm(const float* A, const float* W, const float* bias,
                            float* C, int M, int N, int K, int lda, int ldc,
                            int act, int beta, hipStream_t stream)
{
    dim3 grid((N + 127) / 128, (M + 63) / 64);
    bool ragged = (K % 32) != 0 || (N % 128) != 0;
    if (ragged)
        gemm_bf16_wmma<true><<<grid, 256, 0, stream>>>(A, W, bias, C, M, N, K, lda, ldc, act, beta);
    else
        gemm_bf16_wmma<false><<<grid, 256, 0, stream>>>(A, W, bias, C, M, N, K, lda, ldc, act, beta);
}

extern "C" void kernel_launch(void* const* d_in, const int* in_sizes, int n_in,
                              void* d_out, int out_size, void* d_ws, size_t ws_size,
                              hipStream_t stream)
{
    const int*   idx       = (const int*)  d_in[0];
    const int*   targets   = (const int*)  d_in[1];
    const float* tok_emb   = (const float*)d_in[2];
    const float* pos_emb   = (const float*)d_in[3];
    const float* ln1_g     = (const float*)d_in[4];
    const float* ln1_b     = (const float*)d_in[5];
    const float* in_proj_w = (const float*)d_in[6];
    const float* conv_w    = (const float*)d_in[7];
    const float* conv_b    = (const float*)d_in[8];
    const float* x_proj_w  = (const float*)d_in[9];
    const float* dt_proj_w = (const float*)d_in[10];
    const float* dt_proj_b = (const float*)d_in[11];
    const float* A_log     = (const float*)d_in[12];
    const float* D_skip    = (const float*)d_in[13];
    const float* out_proj_w= (const float*)d_in[14];
    const float* ln2_g     = (const float*)d_in[15];
    const float* ln2_b     = (const float*)d_in[16];
    const float* ffn_w1    = (const float*)d_in[17];
    const float* ffn_b1    = (const float*)d_in[18];
    const float* ffn_w2    = (const float*)d_in[19];
    const float* ffn_b2    = (const float*)d_in[20];
    const float* lnf_g     = (const float*)d_in[21];
    const float* lnf_b     = (const float*)d_in[22];
    const float* head_w    = (const float*)d_in[23];
    const float* head_b    = (const float*)d_in[24];

    // workspace layout (floats)
    float* ws = (float*)d_ws;
    size_t off = 0;
    float* xres = ws + off; off += (size_t)ROWS * Dd;        // residual stream
    float* hln  = ws + off; off += (size_t)ROWS * Dd;        // layernorm out
    float* xzb  = ws + off; off += (size_t)ROWS * 2 * DIi;   // xz / ffn hidden
    float* xcb  = ws + off; off += (size_t)ROWS * DIi;       // conv+silu out
    float* xdbl = ws + off; off += (size_t)ROWS * XPJ;       // dt|B|C
    float* dtb  = ws + off; off += (size_t)ROWS * DIi;       // softplus dt
    float* yb   = ws + off; off += (size_t)ROWS * DIi;       // scan out
    float* rloss= ws + off; off += (size_t)ROWS;             // per-row loss
    (void)ws_size; (void)n_in; (void)in_sizes; (void)out_size;

    float* logits = (float*)d_out;
    float* loss   = logits + (size_t)ROWS * Vv;

    // embedding
    embed_kernel<<<(ROWS * Dd + 255) / 256, 256, 0, stream>>>(idx, tok_emb, pos_emb, xres);

    for (int l = 0; l < NLl; ++l) {
        // --- mamba block ---
        layernorm_kernel<<<ROWS / 8, 256, 0, stream>>>(xres, ln1_g + l * Dd, ln1_b + l * Dd, hln);
        run_gemm(hln, in_proj_w + (size_t)l * 2 * DIi * Dd, nullptr, xzb,
                 ROWS, 2 * DIi, Dd, Dd, 2 * DIi, 0, 0, stream);
        conv_silu_kernel<<<(ROWS * DIi + 255) / 256, 256, 0, stream>>>(
            xzb, conv_w + (size_t)l * DIi * DCc, conv_b + (size_t)l * DIi, xcb);
        run_gemm(xcb, x_proj_w + (size_t)l * XPJ * DIi, nullptr, xdbl,
                 ROWS, XPJ, DIi, DIi, XPJ, 0, 0, stream);
        run_gemm(xdbl, dt_proj_w + (size_t)l * DIi * DTRr, dt_proj_b + (size_t)l * DIi, dtb,
                 ROWS, DIi, DTRr, XPJ, DIi, /*softplus*/2, 0, stream);
        {
            dim3 g(DIi / 256, Bb);
            scan_kernel<<<g, 256, 0, stream>>>(xzb, xcb, xdbl, dtb,
                A_log + (size_t)l * DIi * DSs, D_skip + (size_t)l * DIi, yb);
        }
        run_gemm(yb, out_proj_w + (size_t)l * Dd * DIi, nullptr, xres,
                 ROWS, Dd, DIi, DIi, Dd, 0, /*beta=*/1, stream);

        // --- ffn block ---
        layernorm_kernel<<<ROWS / 8, 256, 0, stream>>>(xres, ln2_g + l * Dd, ln2_b + l * Dd, hln);
        run_gemm(hln, ffn_w1 + (size_t)l * DFFf * Dd, ffn_b1 + (size_t)l * DFFf, xzb,
                 ROWS, DFFf, Dd, Dd, DFFf, /*relu*/1, 0, stream);
        run_gemm(xzb, ffn_w2 + (size_t)l * Dd * DFFf, ffn_b2 + (size_t)l * Dd, xres,
                 ROWS, Dd, DFFf, DFFf, Dd, 0, /*beta=*/1, stream);
    }

    // final LN + head
    layernorm_kernel<<<ROWS / 8, 256, 0, stream>>>(xres, lnf_g, lnf_b, hln);
    run_gemm(hln, head_w, head_b, logits, ROWS, Vv, Dd, Dd, Vv, 0, 0, stream);

    // loss
    row_loss_kernel<<<ROWS / 8, 256, 0, stream>>>(logits, targets, rloss);
    reduce_loss_kernel<<<1, 256, 0, stream>>>(rloss, loss);
}
